// AE_rnn_40364102648137
// MI455X (gfx1250) — compile-verified, matrix operand order checked by reading, not run on hardware
//
#include <hip/hip_runtime.h>

typedef __attribute__((ext_vector_type(16))) _Float16 v16h;
typedef __attribute__((ext_vector_type(8)))  float    v8f;

#define L_SEQ   2048
#define VOCAB   32000
#define IN_DIM  1024
#define H       13
#define G4      52          // 4*H
#define H2      26          // 2*H
#define KPAD    32          // padded K for WMMA (>=26)

__device__ __forceinline__ float sigm(float v) { return 1.0f / (1.0f + __expf(-v)); }

// ---------------- K1: embedding gather: embs[t][:] = emb[x[t]][:] ----------------
__global__ void k_embed(const int* __restrict__ x, const float* __restrict__ emb,
                        float* __restrict__ embs) {
    int t = blockIdx.x;
    long row = (long)x[t];
    const float* src = emb + row * IN_DIM;
    float* dst = embs + (long)t * IN_DIM;
    for (int i = threadIdx.x; i < IN_DIM; i += blockDim.x) dst[i] = src[i];
}

// ---------- K2: layer-0 gate preactivations  G0[d][t][j] = Wih_d[j,:]·embs[t] + bih+bhh ----------
__global__ void k_l0_gates(const float* __restrict__ embs,
                           const float* __restrict__ Wf, const float* __restrict__ bif, const float* __restrict__ bhf,
                           const float* __restrict__ Wb, const float* __restrict__ bib, const float* __restrict__ bhb,
                           float* __restrict__ G0) {
    __shared__ float xrow[IN_DIM];
    int t = blockIdx.x, d = blockIdx.y;
    const float* src = embs + (long)t * IN_DIM;
    for (int i = threadIdx.x; i < IN_DIM; i += blockDim.x) xrow[i] = src[i];
    __syncthreads();
    int j = threadIdx.x;
    if (j < G4) {
        const float* W  = d ? Wb  : Wf;
        const float* bi = d ? bib : bif;
        const float* bh = d ? bhb : bhf;
        const float* wr = W + (long)j * IN_DIM;
        float acc = bi[j] + bh[j];
        #pragma unroll 8
        for (int k = 0; k < IN_DIM; ++k) acc += wr[k] * xrow[k];
        G0[((long)d * L_SEQ + t) * G4 + j] = acc;
    }
}

// ---------- K3: layer-0 sequential scan (one block per direction) ----------
__global__ void k_scan0(const float* __restrict__ G0,
                        const float* __restrict__ WhhF, const float* __restrict__ WhhB,
                        float* __restrict__ Y0, float* __restrict__ hflat, float* __restrict__ cflat) {
    __shared__ float h[H], c[H], g[G4];
    int d = blockIdx.x;
    const float* Whh = d ? WhhB : WhhF;
    int j = threadIdx.x;
    if (j < H) { h[j] = 0.f; c[j] = 0.f; }
    __syncthreads();
    for (int s = 0; s < L_SEQ; ++s) {
        int t = d ? (L_SEQ - 1 - s) : s;
        if (j < G4) {
            float acc = G0[((long)d * L_SEQ + t) * G4 + j];
            const float* wr = Whh + j * H;
            #pragma unroll
            for (int i = 0; i < H; ++i) acc += wr[i] * h[i];
            g[j] = acc;
        }
        __syncthreads();
        if (j < H) {
            float ig = sigm(g[j]);
            float fg = sigm(g[j + H]);
            float gg = tanhf(g[j + 2 * H]);
            float og = sigm(g[j + 3 * H]);
            float c2 = fg * c[j] + ig * gg;
            float h2 = og * tanhf(c2);
            c[j] = c2; h[j] = h2;
            Y0[(long)t * H2 + d * H + j] = h2;
        }
        __syncthreads();
    }
    if (j < H) { hflat[d * H + j] = h[j]; cflat[d * H + j] = c[j]; }
}

// ---------- K4: layer-1 sequential scan (only final h/c needed) ----------
__global__ void k_scan1(const float* __restrict__ Y0,
                        const float* __restrict__ WihF, const float* __restrict__ WhhF,
                        const float* __restrict__ biF,  const float* __restrict__ bhF,
                        const float* __restrict__ WihB, const float* __restrict__ WhhB,
                        const float* __restrict__ biB,  const float* __restrict__ bhB,
                        float* __restrict__ hflat, float* __restrict__ cflat) {
    __shared__ float h[H], c[H], g[G4], xr[H2];
    int d = blockIdx.x;
    const float* Wih = d ? WihB : WihF;
    const float* Whh = d ? WhhB : WhhF;
    const float* bi  = d ? biB  : biF;
    const float* bh  = d ? bhB  : bhF;
    int j = threadIdx.x;
    if (j < H) { h[j] = 0.f; c[j] = 0.f; }
    __syncthreads();
    for (int s = 0; s < L_SEQ; ++s) {
        int t = d ? (L_SEQ - 1 - s) : s;
        if (j < H2) xr[j] = Y0[(long)t * H2 + j];
        __syncthreads();
        if (j < G4) {
            float acc = bi[j] + bh[j];
            const float* wi = Wih + j * H2;
            #pragma unroll
            for (int k = 0; k < H2; ++k) acc += wi[k] * xr[k];
            const float* wr = Whh + j * H;
            #pragma unroll
            for (int i = 0; i < H; ++i) acc += wr[i] * h[i];
            g[j] = acc;
        }
        __syncthreads();
        if (j < H) {
            float ig = sigm(g[j]);
            float fg = sigm(g[j + H]);
            float gg = tanhf(g[j + 2 * H]);
            float og = sigm(g[j + 3 * H]);
            float c2 = fg * c[j] + ig * gg;
            c[j] = c2; h[j] = og * tanhf(c2);
        }
        __syncthreads();
    }
    if (j < H) { hflat[H2 + d * H + j] = h[j]; cflat[H2 + d * H + j] = c[j]; }
}

// ---------- K5: projections + decoder step 0 ----------
__global__ void k_proj_step0(const float* __restrict__ hflat, const float* __restrict__ cflat,
                             const float* __restrict__ p1w, const float* __restrict__ p1b,
                             const float* __restrict__ p2w, const float* __restrict__ p2b,
                             const float* __restrict__ dW0, const float* __restrict__ dWh0,
                             const float* __restrict__ dbi0, const float* __restrict__ dbh0,
                             const float* __restrict__ dW1, const float* __restrict__ dWh1,
                             const float* __restrict__ dbi1, const float* __restrict__ dbh1,
                             float* __restrict__ dech) {
    __shared__ float sh[H2], sc[H2], g[G4];
    int j = threadIdx.x;
    if (j < H2) {
        float a = p1b[j], b = p2b[j];
        for (int k = 0; k < G4; ++k) {
            a += p1w[j * G4 + k] * hflat[k];
            b += p2w[j * G4 + k] * cflat[k];
        }
        sh[j] = a; sc[j] = b;
    }
    __syncthreads();
    for (int d = 0; d < 2; ++d) {
        const float* Wih = d ? dW1  : dW0;
        const float* Whh = d ? dWh1 : dWh0;
        const float* bi  = d ? dbi1 : dbi0;
        const float* bh  = d ? dbh1 : dbh0;
        if (j < G4) {
            float acc = -1.0f * Wih[j] + bi[j] + bh[j];   // x0 = -1.0, Wih is (4H,1)
            for (int i = 0; i < H; ++i) acc += Whh[j * H + i] * sh[d * H + i];
            g[j] = acc;
        }
        __syncthreads();
        if (j < H) {
            float ig = sigm(g[j]);
            float fg = sigm(g[j + H]);
            float gg = tanhf(g[j + 2 * H]);
            float og = sigm(g[j + 3 * H]);
            float c2 = fg * sc[d * H + j] + ig * gg;
            dech[d * H + j] = og * tanhf(c2);             // dec_h row 0
        }
        __syncthreads();
    }
}

// ---------- K6: decoder steps 1..L-1 (zero state => elementwise) ----------
__global__ void k_dec_rest(const int* __restrict__ x,
                           const float* __restrict__ dW0, const float* __restrict__ dbi0, const float* __restrict__ dbh0,
                           const float* __restrict__ dW1, const float* __restrict__ dbi1, const float* __restrict__ dbh1,
                           float* __restrict__ dech) {
    int idx = blockIdx.x * blockDim.x + threadIdx.x;
    const int n = (L_SEQ - 1) * H2;
    if (idx >= n) return;
    int t   = 1 + idx / H2;
    int col = idx % H2;
    int d = col / H;
    int i = col % H;
    const float* Wih = d ? dW1  : dW0;
    const float* bi  = d ? dbi1 : dbi0;
    const float* bh  = d ? dbh1 : dbh0;
    float S = (float)x[t - 1];
    float gi = S * Wih[i]          + bi[i]          + bh[i];
    float gg = S * Wih[i + 2 * H]  + bi[i + 2 * H]  + bh[i + 2 * H];
    float go = S * Wih[i + 3 * H]  + bi[i + 3 * H]  + bh[i + 3 * H];
    float c  = sigm(gi) * tanhf(gg);                 // f*c0 == 0
    dech[(long)t * H2 + col] = sigm(go) * tanhf(c);
}

// ---------- K7: pack dec_h and out_w into K-padded f16 matrices ----------
__global__ void k_packA(const float* __restrict__ dech, _Float16* __restrict__ A16) {
    int idx = blockIdx.x * blockDim.x + threadIdx.x;
    if (idx >= L_SEQ * KPAD) return;
    int m = idx / KPAD, k = idx % KPAD;
    A16[idx] = (k < H2) ? (_Float16)dech[(long)m * H2 + k] : (_Float16)0.0f;
}
__global__ void k_packB(const float* __restrict__ out_w, _Float16* __restrict__ B16) {
    long idx = (long)blockIdx.x * blockDim.x + threadIdx.x;
    if (idx >= (long)VOCAB * KPAD) return;
    long n = idx / KPAD; int k = (int)(idx % KPAD);
    B16[idx] = (k < H2) ? (_Float16)out_w[n * H2 + k] : (_Float16)0.0f;
}

// ---------- K8: logits = dec_h @ out_w^T + out_b via v_wmma_f32_16x16x32_f16 ----------
// One wave computes one 16x16 tile. 128 M-tiles x 2000 N-tiles = 256000 waves.
__global__ void k_logits(const _Float16* __restrict__ A16, const _Float16* __restrict__ B16,
                         const float* __restrict__ out_b, float* __restrict__ logits) {
    int wid   = blockIdx.x * (blockDim.x >> 5) + (threadIdx.x >> 5);
    int lane  = threadIdx.x & 31;
    int mtile = wid & 127;          // 2048/16
    int ntile = wid >> 7;           // 32000/16
    int half  = lane >> 4;
    int l15   = lane & 15;
    long m = (long)(mtile << 4) + l15;
    long n = (long)(ntile << 4) + l15;

    // CDNA5 16-bit 16x32 A/B VGPR striping: lanes 0-15 hold K {0..7,16..23},
    // lanes 16-31 hold K {8..15,24..31}; VGPR j packs K pair at
    // k = ((j&4)<<2) + ((j&3)<<1) + (half<<3).
    v16h a, b;
    #pragma unroll
    for (int j = 0; j < 8; ++j) {
        int k = ((j & 4) << 2) + ((j & 3) << 1) + (half << 3);
        a[2 * j]     = A16[m * KPAD + k];
        a[2 * j + 1] = A16[m * KPAD + k + 1];
        b[2 * j]     = B16[n * KPAD + k];
        b[2 * j + 1] = B16[n * KPAD + k + 1];
    }
    v8f cacc = {};
    cacc = __builtin_amdgcn_wmma_f32_16x16x32_f16(false, a, false, b, (short)0, cacc, false, false);

    float bias = out_b[n];
    #pragma unroll
    for (int r = 0; r < 8; ++r) {
        long row = (long)(mtile << 4) + r + (half << 3);   // C/D: VGPR r -> M=r (+8 for upper half)
        logits[row * VOCAB + (long)(ntile << 4) + l15] = cacc[r] + bias;
    }
}

extern "C" void kernel_launch(void* const* d_in, const int* in_sizes, int n_in,
                              void* d_out, int out_size, void* d_ws, size_t ws_size,
                              hipStream_t stream) {
    (void)in_sizes; (void)n_in; (void)out_size; (void)ws_size;
    const int*   x     = (const int*)d_in[0];
    const float* emb   = (const float*)d_in[1];
    // enc_params leaves: layer0 fwd/bwd, layer1 fwd/bwd, each (Wih, Whh, bih, bhh)
    const float* Wih0f = (const float*)d_in[2],  *Whh0f = (const float*)d_in[3];
    const float* bih0f = (const float*)d_in[4],  *bhh0f = (const float*)d_in[5];
    const float* Wih0b = (const float*)d_in[6],  *Whh0b = (const float*)d_in[7];
    const float* bih0b = (const float*)d_in[8],  *bhh0b = (const float*)d_in[9];
    const float* Wih1f = (const float*)d_in[10], *Whh1f = (const float*)d_in[11];
    const float* bih1f = (const float*)d_in[12], *bhh1f = (const float*)d_in[13];
    const float* Wih1b = (const float*)d_in[14], *Whh1b = (const float*)d_in[15];
    const float* bih1b = (const float*)d_in[16], *bhh1b = (const float*)d_in[17];
    const float* p1w = (const float*)d_in[18], *p1b = (const float*)d_in[19];
    const float* p2w = (const float*)d_in[20], *p2b = (const float*)d_in[21];
    const float* dW0 = (const float*)d_in[22], *dWh0 = (const float*)d_in[23];
    const float* dbi0 = (const float*)d_in[24], *dbh0 = (const float*)d_in[25];
    const float* dW1 = (const float*)d_in[26], *dWh1 = (const float*)d_in[27];
    const float* dbi1 = (const float*)d_in[28], *dbh1 = (const float*)d_in[29];
    const float* out_w = (const float*)d_in[30], *out_b = (const float*)d_in[31];

    float* logits = (float*)d_out;
    float* embs   = logits + (size_t)L_SEQ * VOCAB;     // second tuple output

    // workspace layout
    float* G0    = (float*)d_ws;                        // 2*2048*52
    float* Y0    = G0 + 2 * L_SEQ * G4;                 // 2048*26
    float* hflat = Y0 + L_SEQ * H2;                     // 52 (pad 64)
    float* cflat = hflat + 64;                          // 52 (pad 64)
    float* dech  = cflat + 64;                          // 2048*26
    _Float16* A16 = (_Float16*)(dech + L_SEQ * H2);     // 2048*32 halves
    _Float16* B16 = A16 + (size_t)L_SEQ * KPAD;         // 32000*32 halves

    k_embed<<<L_SEQ, 256, 0, stream>>>(x, emb, embs);
    k_l0_gates<<<dim3(L_SEQ, 2), 64, 0, stream>>>(embs, Wih0f, bih0f, bhh0f,
                                                  Wih0b, bih0b, bhh0b, G0);
    k_scan0<<<2, 64, 0, stream>>>(G0, Whh0f, Whh0b, Y0, hflat, cflat);
    k_scan1<<<2, 64, 0, stream>>>(Y0, Wih1f, Whh1f, bih1f, bhh1f,
                                  Wih1b, Whh1b, bih1b, bhh1b, hflat, cflat);
    k_proj_step0<<<1, 64, 0, stream>>>(hflat, cflat, p1w, p1b, p2w, p2b,
                                       dW0, dWh0, dbi0, dbh0, dW1, dWh1, dbi1, dbh1, dech);
    k_dec_rest<<<((L_SEQ - 1) * H2 + 255) / 256, 256, 0, stream>>>(x, dW0, dbi0, dbh0,
                                                                   dW1, dbi1, dbh1, dech);
    k_packA<<<(L_SEQ * KPAD + 255) / 256, 256, 0, stream>>>(dech, A16);
    k_packB<<<(VOCAB * KPAD + 255) / 256, 256, 0, stream>>>(out_w, B16);
    // 128*2000 tiles, 8 waves per 256-thread block -> 32000 blocks, EXEC all-1s everywhere
    k_logits<<<32000, 256, 0, stream>>>(A16, B16, out_b, logits);
}
